// ChebConv_with_sa_10101763080584
// MI455X (gfx1250) — matrix operand order repaired and checked by reading, not compiled, not run on hardware
//
#include <hip/hip_runtime.h>

// ---------------------------------------------------------------------------
// ChebConv with attention mask, MI455X (gfx1250, wave32, WMMA + TDM).
//   out[t,b,m,o] = relu( sum_{k,n,f} cheb[k,n,m]*s_a[b,n,m]*X[b,n,f,t]*Theta[k,f,o] )
// B=16 N=512 F_IN=32 T=24 K=3 C_OUT=64
//
// Stage 1 (dominant, 19.3 GFLOP): Z[k,b,m,ft] = sum_n (cheb*s_a)[k,b,n,m] * X[b,n,ft]
//   - fp32 tiles staged into LDS by the Tensor Data Mover (tensor_load_to_lds),
//     double buffered, TENSORcnt-pipelined one K-step ahead.
//   - LDS fp32 -> (cheb*s_a) -> bf16 fragment-layout tiles (packed b128 stores)
//   - v_wmma_f32_16x16x32_bf16, fp32 accumulation.
// Stage 2 (repack):  Z2[b,t,m,kf] = Z[k,b,m,f*24+t]
// Stage 3 (2.4 GFLOP): out[t,b,m,o] = relu(sum_kf Z2 * Theta_bf16), fused ReLU.
// ---------------------------------------------------------------------------

typedef __attribute__((ext_vector_type(16))) __bf16    v16bf;
typedef __attribute__((ext_vector_type(8)))  float     v8f;
typedef __attribute__((ext_vector_type(4)))  unsigned  u32x4;
typedef __attribute__((ext_vector_type(4)))  int       i32x4;
typedef __attribute__((ext_vector_type(8)))  int       i32x8;

union FragBf {
    v16bf v;
    u32x4 q[2];
};

union Pack8 {
    u32x4 q;
    unsigned short h[8];
};

__device__ __forceinline__ unsigned short f2bf(float f) {
    unsigned u = __builtin_bit_cast(unsigned, f);
    unsigned r = u + 0x7FFFu + ((u >> 16) & 1u);   // round-to-nearest-even
    return (unsigned short)(r >> 16);
}

#define NN   512
#define FT   768      // F_IN * T
#define KF   96       // K * F_IN
#define BM   64       // m-tile
#define BFT  64       // ft-tile
#define BK   32       // n step (bf16 WMMA K)

// ---------------------------------------------------------------------------
// TDM: issue one 2D-tile Global->LDS DMA (fp32 elements).
// D# group0/group1 per CDNA5 ISA 8.3/8.4; VADDR2/3 = NULL (2D tensor).
// dims/strides in elements (data_size = 4 bytes).
// ---------------------------------------------------------------------------
__device__ __forceinline__ void tdm_load_tile_f32(
    unsigned lds_off, const float* gaddr,
    unsigned tensor_d0, unsigned tensor_d1, unsigned stride0,
    unsigned tile_d0, unsigned tile_d1)
{
    unsigned long long ga = (unsigned long long)(size_t)gaddr;
    i32x4 g0;
    g0[0] = 1;                                     // count=1, user descriptor
    g0[1] = (int)lds_off;                          // lds_addr
    g0[2] = (int)(unsigned)ga;                     // global_addr[31:0]
    g0[3] = (int)((unsigned)((ga >> 32) & 0x01FFFFFFu) | 0x80000000u); // addr[56:32] | type=2
    i32x8 g1;
    g1[0] = (int)(2u << 16);                       // workgroup_mask=0, data_size=2 (4B)
    g1[1] = (int)(tensor_d0 << 16);                // [63:48] = tensor_dim0[15:0]
    g1[2] = (int)((tensor_d0 >> 16) | (tensor_d1 << 16));
    g1[3] = (int)((tensor_d1 >> 16) | (tile_d0 << 16));
    g1[4] = (int)tile_d1;                          // tile_dim1, tile_dim2=0
    g1[5] = (int)stride0;                          // tensor_dim0_stride[31:0]
    g1[6] = 0;                                     // stride0[47:32]=0, stride1[15:0]=0
    g1[7] = 0;
    asm volatile("tensor_load_to_lds %0, %1"
                 :: "s"(g0), "s"(g1)
                 : "memory");
}

// ---------------------------------------------------------------------------
// Kernel 1: Z[k][b][m][ft] (bf16) = sum_n (cheb[k][n][m]*s_a[b][n][m]) * X[b][n][ft]
// grid = (FT/64, 512/64, K*B), block = 256 (8 waves)
// ---------------------------------------------------------------------------
__global__ __launch_bounds__(256) void gemm1_kernel(
    const float* __restrict__ X, const float* __restrict__ s_a,
    const float* __restrict__ cheb, unsigned short* __restrict__ Z)
{
    // TDM staging (double buffered): raw fp32 tiles, 32 rows x 64 cols each
    __shared__ __align__(16) float stC[2][BK * BM];   // cheb
    __shared__ __align__(16) float stS[2][BK * BM];   // s_a
    __shared__ __align__(16) float stX[2][BK * BFT];  // X
    // A-frag friendly: As[m][n], row stride 40 halves (80B, 16B aligned)
    __shared__ __align__(16) unsigned short As[BM][40];
    // B-frag friendly: Bs[n/16][ft][n%16], inner padded to 24 halves (48B)
    __shared__ __align__(16) unsigned short Bs[2][BFT][24];

    const int ftb    = blockIdx.x * BFT;
    const int mb     = blockIdx.y * BM;
    const int kb_lin = blockIdx.z;          // 0..47
    const int k = kb_lin % 3;
    const int b = kb_lin / 3;

    const float* __restrict__ chebk = cheb + (size_t)k * NN * NN;
    const float* __restrict__ sab   = s_a  + (size_t)b * NN * NN;
    const float* __restrict__ Xb    = X    + (size_t)b * NN * FT;

    const int tid  = threadIdx.x;
    const int lane = tid & 31;
    const int w    = tid >> 5;              // wave 0..7
    const int mi   = (w & 3) * 16;          // m subtile in block
    const int fti0 = (w >> 2) * 32;         // ft subtile pair base

    const int cm    = tid & 63;             // column handled by this thread
    const int nbase = (tid >> 6) * 8;       // 8 consecutive n rows per thread

    const unsigned offC0 = (unsigned)(size_t)&stC[0][0];
    const unsigned offC1 = (unsigned)(size_t)&stC[1][0];
    const unsigned offS0 = (unsigned)(size_t)&stS[0][0];
    const unsigned offS1 = (unsigned)(size_t)&stS[1][0];
    const unsigned offX0 = (unsigned)(size_t)&stX[0][0];
    const unsigned offX1 = (unsigned)(size_t)&stX[1][0];

    v8f acc0 = {};
    v8f acc1 = {};

    // Prologue: wave 0 launches the TDM for K-step 0 into buffer 0.
    if (w == 0) {
        tdm_load_tile_f32(offC0, chebk + mb,  NN, NN, NN, BM, BK);
        tdm_load_tile_f32(offS0, sab + mb,    NN, NN, NN, BM, BK);
        tdm_load_tile_f32(offX0, Xb + ftb,    FT, NN, FT, BFT, BK);
    }

    const int NSTEP = NN / BK;              // 16
    for (int i = 0; i < NSTEP; ++i) {
        const int cur = i & 1;
        if (w == 0) {
            if (i + 1 < NSTEP) {            // issue next K-step tiles (other buffer)
                const int n1 = (i + 1) * BK;
                tdm_load_tile_f32(cur ? offC0 : offC1, chebk + (size_t)n1 * NN + mb,
                                  NN, NN, NN, BM, BK);
                tdm_load_tile_f32(cur ? offS0 : offS1, sab + (size_t)n1 * NN + mb,
                                  NN, NN, NN, BM, BK);
                tdm_load_tile_f32(cur ? offX0 : offX1, Xb + (size_t)n1 * FT + ftb,
                                  FT, NN, FT, BFT, BK);
                // TDM completes in order per wave: <=3 outstanding means the
                // current step's three tiles have landed in LDS.
                __builtin_amdgcn_s_wait_tensorcnt(3);
            } else {
                __builtin_amdgcn_s_wait_tensorcnt(0);
            }
        }
        __syncthreads();   // publish TDM data; previous frag reads are done too

        // --- convert: fp32 staging -> bf16 frag tiles, packed b128 stores ---
        const float* __restrict__ pc = &stC[cur][0];
        const float* __restrict__ ps = &stS[cur][0];
        const float* __restrict__ px = &stX[cur][0];
        Pack8 va, vb;
        #pragma unroll
        for (int r = 0; r < 8; ++r) {
            const int nn = nbase + r;       // 8 consecutive rows
            va.h[r] = f2bf(pc[nn * BM + cm] * ps[nn * BM + cm]);
            vb.h[r] = f2bf(px[nn * BFT + cm]);
        }
        *(u32x4*)&As[cm][nbase] = va.q;                       // 16B aligned
        *(u32x4*)&Bs[nbase >> 4][cm][nbase & 15] = vb.q;      // 16B aligned
        __syncthreads();

        // --- fragment loads (contiguous k = 16*(lane/16) + s, A and B alike, wave32) ---
        FragBf fa, fb0, fb1;
        {
            const int m   = mi + (lane & 15);
            const int kb0 = (lane < 16) ? 0 : 16;
            const unsigned short* rp = &As[m][0];
            fa.q[0] = *(const u32x4*)(rp + kb0);        // K = kb0..kb0+7
            fa.q[1] = *(const u32x4*)(rp + 8 + kb0);    // K = kb0+8..kb0+15
        }
        {
            const int g  = (lane < 16) ? 0 : 1;         // K group 0-15 / 16-31
            const int c0 = fti0 + (lane & 15);
            const unsigned short* p0 = &Bs[g][c0][0];
            fb0.q[0] = *(const u32x4*)(p0);
            fb0.q[1] = *(const u32x4*)(p0 + 8);
            const unsigned short* p1 = &Bs[g][c0 + 16][0];
            fb1.q[0] = *(const u32x4*)(p1);
            fb1.q[1] = *(const u32x4*)(p1 + 8);
        }
        acc0 = __builtin_amdgcn_wmma_f32_16x16x32_bf16(
                   false, fa.v, false, fb0.v, (short)0, acc0, false, false);
        acc1 = __builtin_amdgcn_wmma_f32_16x16x32_bf16(
                   false, fa.v, false, fb1.v, (short)0, acc1, false, false);
    }

    // --- epilogue: store Z tile (bf16) ---
    unsigned short* __restrict__ Zp = Z + (size_t)kb_lin * NN * FT;
    const int col0  = ftb + fti0 + (lane & 15);
    const int rbase = mb + mi + ((lane >> 4) << 3);     // lanes 16-31 -> rows +8
    #pragma unroll
    for (int i = 0; i < 8; ++i) {
        const size_t m = (size_t)(rbase + i);
        Zp[m * FT + col0]      = f2bf(acc0[i]);
        Zp[m * FT + col0 + 16] = f2bf(acc1[i]);
    }
}

// ---------------------------------------------------------------------------
// Kernel 2: repack  Z[k][b][m][f*24+t]  ->  Z2[b][t][m][k*32+f]
// grid = (512/8, 16), block = 256
// ---------------------------------------------------------------------------
__global__ __launch_bounds__(256) void repack_kernel(
    const unsigned short* __restrict__ Z, unsigned short* __restrict__ Z2)
{
    __shared__ __align__(16) unsigned short S[3][8][FT];   // 36 KB

    const int m0 = blockIdx.x * 8;
    const int b  = blockIdx.y;
    const int tid = threadIdx.x;

    // load: 3 k-slices x 8 m-rows x 768 halves, 16B vectors (96 per row)
    for (int idx = tid; idx < 3 * 8 * 96; idx += 256) {
        const int k   = idx / (8 * 96);
        const int rem = idx % (8 * 96);
        const int mm  = rem / 96;
        const int c   = rem % 96;
        const u32x4* src = (const u32x4*)
            (Z + ((size_t)(k * 16 + b) * NN + m0 + mm) * FT);
        ((u32x4*)&S[k][mm][0])[c] = src[c];
    }
    __syncthreads();

    for (int idx = tid; idx < 8 * 24 * KF; idx += 256) {
        const int mm  = idx / (24 * KF);
        const int rem = idx % (24 * KF);
        const int t   = rem / KF;
        const int kf  = rem % KF;
        const int k = kf >> 5, f = kf & 31;
        Z2[((size_t)(b * 24 + t) * NN + m0 + mm) * KF + kf] = S[k][mm][f * 24 + t];
    }
}

// ---------------------------------------------------------------------------
// Kernel 3: out[t][b][m][o] = relu( sum_kf Z2[b][t][m][kf] * Theta_bf16[kf][o] )
// grid = (512/64, 24, 16), block = 256 (8 waves)
// ---------------------------------------------------------------------------
__global__ __launch_bounds__(256) void gemm2_kernel(
    const unsigned short* __restrict__ Z2, const float* __restrict__ Theta,
    float* __restrict__ out)
{
    __shared__ __align__(16) unsigned short As2[64][104];   // m x 96 (pad -> 208B rows)
    __shared__ __align__(16) unsigned short Ts[3][2][64][24];

    const int m0 = blockIdx.x * 64;
    const int t  = blockIdx.y;
    const int b  = blockIdx.z;
    const int tid  = threadIdx.x;
    const int lane = tid & 31;
    const int w    = tid >> 5;

    // stage Z2 tile: 64 rows x 96 halves (contiguous), 16B vectors (12 per row)
    const u32x4* src = (const u32x4*)
        (Z2 + ((size_t)(b * 24 + t) * NN + m0) * KF);
    for (int idx = tid; idx < 64 * 12; idx += 256) {
        const int m = idx / 12, c = idx % 12;
        *(u32x4*)&As2[m][c * 8] = src[m * 12 + c];
    }
    // stage Theta (3*32*64 f32) as bf16 in B-frag layout: thread per (k,g,o)
    for (int idx = tid; idx < 3 * 2 * 64; idx += 256) {
        const int k   = idx >> 7;
        const int rem = idx & 127;
        const int g = rem >> 6;
        const int o = rem & 63;
        Pack8 lo, hi;
        #pragma unroll
        for (int fp = 0; fp < 8; ++fp) {
            lo.h[fp] = f2bf(Theta[(k * 32 + g * 16 + fp) * 64 + o]);
            hi.h[fp] = f2bf(Theta[(k * 32 + g * 16 + 8 + fp) * 64 + o]);
        }
        *(u32x4*)&Ts[k][g][o][0] = lo.q;
        *(u32x4*)&Ts[k][g][o][8] = hi.q;
    }
    __syncthreads();

    const int mi  = (w & 3) * 16;
    const int oi0 = (w >> 2) * 32;
    v8f acc0 = {};
    v8f acc1 = {};

    #pragma unroll
    for (int ks = 0; ks < 3; ++ks) {
        FragBf fa, fb0, fb1;
        const int m   = mi + (lane & 15);
        const int kb0 = (lane < 16) ? 0 : 16;
        const unsigned short* rp = &As2[m][ks * 32];
        fa.q[0] = *(const u32x4*)(rp + kb0);
        fa.q[1] = *(const u32x4*)(rp + 8 + kb0);
        const int g = (lane < 16) ? 0 : 1;
        const unsigned short* p0 = &Ts[ks][g][oi0 + (lane & 15)][0];
        fb0.q[0] = *(const u32x4*)(p0);
        fb0.q[1] = *(const u32x4*)(p0 + 8);
        const unsigned short* p1 = &Ts[ks][g][oi0 + 16 + (lane & 15)][0];
        fb1.q[0] = *(const u32x4*)(p1);
        fb1.q[1] = *(const u32x4*)(p1 + 8);
        acc0 = __builtin_amdgcn_wmma_f32_16x16x32_bf16(
                   false, fa.v, false, fb0.v, (short)0, acc0, false, false);
        acc1 = __builtin_amdgcn_wmma_f32_16x16x32_bf16(
                   false, fa.v, false, fb1.v, (short)0, acc1, false, false);
    }

    // epilogue: fused ReLU, fp32 store, out[t][b][m][o]
    float* __restrict__ op = out + ((size_t)(t * 16 + b) * NN + m0) * 64;
    const int col   = oi0 + (lane & 15);
    const int rbase = mi + ((lane >> 4) << 3);
    #pragma unroll
    for (int i = 0; i < 8; ++i) {
        const float v0 = acc0[i];
        const float v1 = acc1[i];
        op[(size_t)(rbase + i) * 64 + col]      = v0 > 0.f ? v0 : 0.f;
        op[(size_t)(rbase + i) * 64 + col + 16] = v1 > 0.f ? v1 : 0.f;
    }
}

// ---------------------------------------------------------------------------
extern "C" void kernel_launch(void* const* d_in, const int* in_sizes, int n_in,
                              void* d_out, int out_size, void* d_ws, size_t ws_size,
                              hipStream_t stream)
{
    const float* X     = (const float*)d_in[0];   // [16,512,32,24]
    const float* s_a   = (const float*)d_in[1];   // [16,512,512]
    const float* cheb  = (const float*)d_in[2];   // [3,512,512]
    const float* Theta = (const float*)d_in[3];   // [3,32,64]

    unsigned short* Z  = (unsigned short*)d_ws;                 // [3,16,512,768] bf16
    unsigned short* Z2 = Z + (size_t)3 * 16 * NN * FT;          // [16,24,512,96] bf16

    dim3 g1(FT / BFT, NN / BM, 48);
    gemm1_kernel<<<g1, 256, 0, stream>>>(X, s_a, cheb, Z);

    dim3 g2(NN / 8, 16);
    repack_kernel<<<g2, 256, 0, stream>>>(Z, Z2);

    dim3 g3(NN / 64, 24, 16);
    gemm2_kernel<<<g3, 256, 0, stream>>>(Z2, Theta, (float*)d_out);
}
